// ResidualGINLayer_1812476199538
// MI455X (gfx1250) — compile-verified, matrix operand order checked by reading, not run on hardware
//
#include <hip/hip_runtime.h>
#include <hip/hip_bf16.h>

#define D_IN  128
#define D_HID 512
#define BN_EPS 1e-5f
#define LN_EPS 1e-5f

typedef __attribute__((ext_vector_type(16))) __bf16 v16bf;
typedef __attribute__((ext_vector_type(8)))  float  v8f;

union ABFrag { v16bf v; unsigned int u[8]; };

__device__ __forceinline__ unsigned short f2bf(float f) {
  unsigned int u = __float_as_uint(f);
  unsigned int r = u + 0x7FFFu + ((u >> 16) & 1u);   // round-to-nearest-even
  return (unsigned short)(r >> 16);
}

// ---------------- elementwise helpers ----------------

__global__ void k_zero(float* __restrict__ p, int n) {
  int i = blockIdx.x * blockDim.x + threadIdx.x;
  if (i < n) p[i] = 0.0f;
}

__global__ void k_copy_f4(const float4* __restrict__ src, float4* __restrict__ dst, long long n4) {
  long long i = (long long)blockIdx.x * blockDim.x + threadIdx.x;
  if (i < n4) dst[i] = src[i];
}

__global__ void k_f32_to_bf16(const float* __restrict__ src, unsigned short* __restrict__ dst,
                              long long n) {
  long long i = (long long)blockIdx.x * blockDim.x + threadIdx.x;
  if (i < n) dst[i] = f2bf(src[i]);
}

// W: [K, N] row-major fp32  ->  Wt: [N, K] row-major bf16
__global__ void k_w_transpose_bf16(const float* __restrict__ W, unsigned short* __restrict__ Wt,
                                   int K, int N) {
  int t = blockIdx.x * blockDim.x + threadIdx.x;
  if (t >= K * N) return;
  int nIdx = t / K;
  int k    = t - nIdx * K;
  Wt[t] = f2bf(W[(size_t)k * N + nIdx]);
}

// ---------------- GIN aggregation: h[dst] += x[src] (h pre-initialized to x) ----------------

__global__ void k_scatter(const float* __restrict__ x, const long long* __restrict__ ei,
                          float* __restrict__ h, int nE) {
  int e = (int)(((long long)blockIdx.x * blockDim.x + threadIdx.x) >> 5);
  if (e >= nE) return;
  int lane = threadIdx.x & 31;
  long long s = ei[e];
  long long d = ei[(long long)nE + e];
  const float* xs = x + (size_t)s * D_IN + lane * 4;
  float*       hd = h + (size_t)d * D_IN + lane * 4;
#pragma unroll
  for (int j = 0; j < 4; ++j) atomicAdd(hd + j, xs[j]);
}

// ---------------- bf16 WMMA GEMM: C[M,N] = A[M,K] @ Bt[N,K]^T + bias ----------------
// Block = 256 threads = 8 waves. Block covers 128(M) x 64(N):
//   - the shared 64xK B tile (contiguous in Bt) is async-copied to LDS once per block
//     (global_load_async_to_lds_b128 + s_wait_asynccnt + barrier),
//   - each wave computes a 16(M) x 64(N) strip: A fragment reused by 4 WMMAs per k-step,
//     B fragments read from LDS (ds_load_b128).
// Fused BN stats: per-column sum & sumsq accumulated with fp32 atomics
// (lane L and L^16 hold the same output column -> shfl_xor(16) folds the tile).
// 16-bit A/B fragment layout (CDNA5 ISA 7.12.2):
//   lane<16  holds row/col (lane&15), K in {kk+0..7, kk+16..23}
//   lane>=16 holds row/col (lane&15), K in {kk+8..15, kk+24..31}
template <int K>
__global__ void k_gemm_bf16(const unsigned short* __restrict__ A,
                            const unsigned short* __restrict__ Bt,
                            const float* __restrict__ bias,
                            float* __restrict__ C,
                            float* __restrict__ accS, float* __restrict__ accQ,
                            int M, int N) {
  __shared__ unsigned short sB[64 * K];          // 64 rows (N) x K, row-major

  int ntiles = N >> 6;                           // 64-wide N tiles
  int bm = blockIdx.x / ntiles;                  // group of 8 M sub-tiles
  int tn = blockIdx.x - bm * ntiles;

  // ---- stage B tile into LDS with gfx1250 async copies (16B per lane) ----
  const char* Btile = (const char*)(Bt + (size_t)tn * 64 * K);
  unsigned int ldsBase = (unsigned int)(uintptr_t)(&sB[0]);   // low 32b of flat = LDS offset
  constexpr int tileBytes = 64 * K * 2;
  for (int it = threadIdx.x * 16; it < tileBytes; it += 256 * 16) {
    unsigned int ldsOff = ldsBase + (unsigned int)it;
    const char* g = Btile + it;
    asm volatile("global_load_async_to_lds_b128 %0, %1, off"
                 :: "v"(ldsOff), "v"(g) : "memory");
  }
  asm volatile("s_wait_asynccnt 0x0" ::: "memory");
  __syncthreads();

  int wave = threadIdx.x >> 5;
  int tm = bm * 8 + wave;
  if (tm < (M >> 4)) {
    int lane = threadIdx.x & 31;
    int half = lane >> 4;
    int l15  = lane & 15;

    const unsigned short* Arow = A + (size_t)(tm * 16 + l15) * K;

    v8f acc[4] = {{}, {}, {}, {}};
    for (int kk = 0; kk < K; kk += 32) {
      ABFrag a;
#pragma unroll
      for (int p = 0; p < 8; ++p) {
        int k = kk + half * 8 + ((p < 4) ? (2 * p) : (16 + 2 * (p - 4)));
        a.u[p] = *(const unsigned int*)(Arow + k);
      }
      if (kk + 32 < K) __builtin_prefetch(Arow + kk + 32, 0, 0);  // global_prefetch_b8

      ABFrag b[4];
#pragma unroll
      for (int p = 0; p < 8; ++p) {
        int k = kk + half * 8 + ((p < 4) ? (2 * p) : (16 + 2 * (p - 4)));
#pragma unroll
        for (int sub = 0; sub < 4; ++sub)
          b[sub].u[p] = *(const unsigned int*)(&sB[(sub * 16 + l15) * K + k]);
      }
#pragma unroll
      for (int sub = 0; sub < 4; ++sub)
        acc[sub] = __builtin_amdgcn_wmma_f32_16x16x32_bf16(false, a.v, false, b[sub].v,
                                                           (short)0, acc[sub], false, false);
    }

    // epilogue: bias add, store, fused column stats.
    // C/D layout: VGPR v -> row = v + 8*half; lane L and L^16 share the same column.
#pragma unroll
    for (int sub = 0; sub < 4; ++sub) {
      int col = tn * 64 + sub * 16 + l15;
      float bv = bias[col];
      float s = 0.0f, q = 0.0f;
#pragma unroll
      for (int v = 0; v < 8; ++v) {
        int row = tm * 16 + v + 8 * half;
        float y = acc[sub][v] + bv;
        C[(size_t)row * N + col] = y;
        s += y;
        q += y * y;
      }
      s += __shfl_xor(s, 16, 32);
      q += __shfl_xor(q, 16, 32);
      if (half == 0) {
        atomicAdd(&accS[col], s);
        atomicAdd(&accQ[col], q);
      }
    }
  }
}

// ---------------- BatchNorm stats finalize ----------------

__global__ void k_col_stats_final(const float* __restrict__ accS, const float* __restrict__ accQ,
                                  int n, int D, float* __restrict__ mean,
                                  float* __restrict__ invstd) {
  int c = blockIdx.x * blockDim.x + threadIdx.x;
  if (c >= D) return;
  float inv_n = 1.0f / (float)n;
  float m = accS[c] * inv_n;
  float var = accQ[c] * inv_n - m * m;
  mean[c] = m;
  invstd[c] = rsqrtf(fmaxf(var, 0.0f) + BN_EPS);
}

// ---------------- BN + ReLU + convert to bf16 ----------------

__global__ void k_bn_relu_bf16(const float* __restrict__ Y, const float* __restrict__ mean,
                               const float* __restrict__ invstd, const float* __restrict__ g,
                               const float* __restrict__ b, unsigned short* __restrict__ out,
                               long long total, int Dmask) {
  long long i = (long long)blockIdx.x * blockDim.x + threadIdx.x;
  if (i >= total) return;
  int c = (int)(i & Dmask);
  float v = (Y[i] - mean[c]) * invstd[c] * g[c] + b[c];
  out[i] = f2bf(fmaxf(v, 0.0f));
}

// ---------------- final: out = LN( relu(BN2(Y2)) + x ) ----------------
// One wave per row of 128; lane holds 4 values; cross-lane reduce via shfl_xor.
__global__ void k_final_ln(const float* __restrict__ Y2, const float* __restrict__ x,
                           const float* __restrict__ m2, const float* __restrict__ is2,
                           const float* __restrict__ g2, const float* __restrict__ b2,
                           const float* __restrict__ lng, const float* __restrict__ lnb,
                           float* __restrict__ out, int n) {
  int row = (int)(((long long)blockIdx.x * blockDim.x + threadIdx.x) >> 5);
  if (row >= n) return;
  int lane = threadIdx.x & 31;
  const float* yr = Y2 + (size_t)row * D_IN;
  const float* xr = x  + (size_t)row * D_IN;

  float v[4];
  float s = 0.0f;
#pragma unroll
  for (int j = 0; j < 4; ++j) {
    int c = lane * 4 + j;
    float t = (yr[c] - m2[c]) * is2[c] * g2[c] + b2[c];
    t = fmaxf(t, 0.0f) + xr[c];
    v[j] = t;
    s += t;
  }
#pragma unroll
  for (int o = 16; o >= 1; o >>= 1) s += __shfl_xor(s, o, 32);
  float mean = s * (1.0f / 128.0f);

  float q = 0.0f;
#pragma unroll
  for (int j = 0; j < 4; ++j) { float d = v[j] - mean; q += d * d; }
#pragma unroll
  for (int o = 16; o >= 1; o >>= 1) q += __shfl_xor(q, o, 32);
  float inv = rsqrtf(q * (1.0f / 128.0f) + LN_EPS);

  float* orow = out + (size_t)row * D_IN;
#pragma unroll
  for (int j = 0; j < 4; ++j) {
    int c = lane * 4 + j;
    orow[c] = (v[j] - mean) * inv * lng[c] + lnb[c];
  }
}

// ---------------- launcher ----------------

extern "C" void kernel_launch(void* const* d_in, const int* in_sizes, int n_in,
                              void* d_out, int out_size, void* d_ws, size_t ws_size,
                              hipStream_t stream) {
  const float*     x    = (const float*)d_in[0];
  const long long* ei   = (const long long*)d_in[1];
  const float*     W1   = (const float*)d_in[2];
  const float*     b1   = (const float*)d_in[3];
  const float*     bn1g = (const float*)d_in[4];
  const float*     bn1b = (const float*)d_in[5];
  const float*     W2   = (const float*)d_in[6];
  const float*     b2   = (const float*)d_in[7];
  const float*     bn2g = (const float*)d_in[8];
  const float*     bn2b = (const float*)d_in[9];
  const float*     lng  = (const float*)d_in[10];
  const float*     lnb  = (const float*)d_in[11];
  float* out = (float*)d_out;

  int n = in_sizes[0] / D_IN;   // 50000
  int E = in_sizes[1] / 2;      // 800000

  // workspace layout
  char* p = (char*)d_ws;
  float* h  = (float*)p;  p += (size_t)n * D_IN  * 4;
  float* Y1 = (float*)p;  p += (size_t)n * D_HID * 4;
  float* Y2 = (float*)p;  p += (size_t)n * D_IN  * 4;
  unsigned short* Hb  = (unsigned short*)p; p += (size_t)n * D_IN  * 2;
  unsigned short* H1b = (unsigned short*)p; p += (size_t)n * D_HID * 2;
  unsigned short* W1t = (unsigned short*)p; p += (size_t)D_IN * D_HID * 2;
  unsigned short* W2t = (unsigned short*)p; p += (size_t)D_IN * D_HID * 2;
  float* stats = (float*)p;
  float* aS1 = stats;         float* aQ1 = stats + 512;
  float* m1  = stats + 1024;  float* is1 = stats + 1536;
  float* aS2 = stats + 2048;  float* aQ2 = stats + 2176;
  float* m2  = stats + 2304;  float* is2 = stats + 2432;   // total 2560 floats

  // 0) zero BN accumulators
  k_zero<<<(2560 + 255) / 256, 256, 0, stream>>>(stats, 2560);

  // 1) h = x ; h[dst] += x[src] over edges
  long long n4 = (long long)n * (D_IN / 4);
  k_copy_f4<<<(unsigned)((n4 + 255) / 256), 256, 0, stream>>>((const float4*)x, (float4*)h, n4);
  k_scatter<<<(E + 7) / 8, 256, 0, stream>>>(x, ei, h, E);

  // 2) bf16 conversions: activations + pre-transposed weights
  long long nh = (long long)n * D_IN;
  k_f32_to_bf16<<<(unsigned)((nh + 255) / 256), 256, 0, stream>>>(h, Hb, nh);
  k_w_transpose_bf16<<<(D_IN * D_HID + 255) / 256, 256, 0, stream>>>(W1, W1t, D_IN, D_HID);
  k_w_transpose_bf16<<<(D_IN * D_HID + 255) / 256, 256, 0, stream>>>(W2, W2t, D_HID, D_IN);

  int mtiles = n / 16;                 // 3125
  int mblocks = (mtiles + 7) / 8;      // 8 M sub-tiles per block

  // 3) GEMM1 (+fused BN1 stats): Y1 = h @ W1 + b1   [n, 512]
  k_gemm_bf16<D_IN><<<mblocks * (D_HID / 64), 256, 0, stream>>>(Hb, W1t, b1, Y1, aS1, aQ1,
                                                                n, D_HID);

  // 4) BN1 finalize, then BN1+ReLU -> bf16
  k_col_stats_final<<<(D_HID + 127) / 128, 128, 0, stream>>>(aS1, aQ1, n, D_HID, m1, is1);
  long long n1 = (long long)n * D_HID;
  k_bn_relu_bf16<<<(unsigned)((n1 + 255) / 256), 256, 0, stream>>>(Y1, m1, is1, bn1g, bn1b,
                                                                   H1b, n1, D_HID - 1);

  // 5) GEMM2 (+fused BN2 stats): Y2 = h1 @ W2 + b2   [n, 128]
  k_gemm_bf16<D_HID><<<mblocks * (D_IN / 64), 256, 0, stream>>>(H1b, W2t, b2, Y2, aS2, aQ2,
                                                                n, D_IN);

  // 6) BN2 finalize
  k_col_stats_final<<<1, 128, 0, stream>>>(aS2, aQ2, n, D_IN, m2, is2);

  // 7) out = LN( relu(BN2(Y2)) + x )
  k_final_ln<<<(n + 7) / 8, 256, 0, stream>>>(Y2, x, m2, is2, bn2g, bn2b, lng, lnb, out, n);
}
